// DyConvAtten_44693429682893
// MI455X (gfx1250) — matrix operand order repaired and checked by reading, not compiled, not run on hardware
//
#include <hip/hip_runtime.h>
#include <hip/hip_bf16.h>

typedef __attribute__((ext_vector_type(16))) __bf16 v16bf;
typedef __attribute__((ext_vector_type(8)))  float  v8f;

#define HID 256
#define NP  100
#define KS  3
#define MPAD 112            // 7 * 16
#define N1  304             // 19 * 16  (>= NP*KS = 300)
#define K2  320             // 10 * 32  (>= 300), K of GEMM2
#define BATCH 1024
#define EPS 1e-5f

// LDS layout (bytes)
#define OFF_FA    0                          // bf16 [112][256]  = 57344 B
#define OFF_WMAT  57344                      // bf16 [112][320]  = 71680 B
#define OFF_KBT   129024                     // bf16 [256][320]  = 163840 B
#define OFF_SUM   292864                     // f32  [112]
#define OFF_SQ    293312                     // f32  [112]
#define SMEM_BYTES 293760

union FragBf {
    uint4  u[2];
    v16bf  v;
};

// ---------------------------------------------------------------------------
// Prep: W_lin (256x300 f32, row-major) -> wsT (304x256 bf16), wsT[o][h] = W_lin[h][o]
// Rows 300..303 zeroed (N padding for GEMM1).
// ---------------------------------------------------------------------------
__global__ void dyconv_prep_wlinT(const float* __restrict__ W_lin,
                                  __bf16* __restrict__ wsT) {
    int idx = blockIdx.x * blockDim.x + threadIdx.x;  // over 304*256
    if (idx >= N1 * HID) return;
    int o = idx >> 8;
    int h = idx & 255;
    float v = (o < NP * KS) ? W_lin[h * (NP * KS) + o] : 0.0f;
    wsT[idx] = (__bf16)v;
}

// ---------------------------------------------------------------------------
// Fused main kernel: one workgroup (8 wave32) per batch element.
// ---------------------------------------------------------------------------
__global__ void __launch_bounds__(256)
dyconv_fused(const float* __restrict__ f,      // [B,100,256]
             const float* __restrict__ k,      // [B,100,256]
             const float* __restrict__ b_lin,  // [300]
             const float* __restrict__ gamma,  // [256]
             const float* __restrict__ beta,   // [256]
             const __bf16* __restrict__ wsT,   // [304][256] bf16 (W_lin^T)
             float* __restrict__ out)          // [B,100,256]
{
    extern __shared__ char smem[];
    __bf16* fA    = (__bf16*)(smem + OFF_FA);     // [112][256]
    __bf16* Wmat  = (__bf16*)(smem + OFF_WMAT);   // [112][320]
    __bf16* kBt   = (__bf16*)(smem + OFF_KBT);    // [256][320] (B^T of GEMM2)
    float*  sums  = (float*) (smem + OFF_SUM);    // [112]
    float*  sumsq = (float*) (smem + OFF_SQ);     // [112]

    const int b    = blockIdx.x;
    const int tid  = threadIdx.x;
    const int lane = tid & 31;
    const int wave = tid >> 5;
    const int g    = lane >> 4;    // half-wave id (K-group / M-group select)
    const int ln   = lane & 15;

    const float* fb = f + (size_t)b * NP * HID;
    const float* kb = k + (size_t)b * NP * HID;

    // ---- Phase 0a: stage f[b] as bf16 A-matrix [112][256], rows >=100 zero
    {
        const float4* f4 = (const float4*)fb;
        for (int i = tid; i < MPAD * (HID / 4); i += 256) {
            int m  = i >> 6;           // row
            int c4 = i & 63;           // float4 column
            float4 v;
            if (m < NP) v = f4[m * (HID / 4) + c4];
            else        v = make_float4(0.f, 0.f, 0.f, 0.f);
            __bf16* d = &fA[m * HID + c4 * 4];
            d[0] = (__bf16)v.x; d[1] = (__bf16)v.y;
            d[2] = (__bf16)v.z; d[3] = (__bf16)v.w;
        }
    }

    // ---- Phase 0b (zero-fill): K-padding rows r=300..319 for all n (5120 elems)
    for (int i = tid; i < (K2 - NP * KS - 0) * 0 + 20 * HID; i += 256) {
        int n = i & 255;
        int r = NP * KS + (i >> 8);     // 300..319
        kBt[n * K2 + r] = (__bf16)0.0f;
    }
    // Boundary entries not covered by the scatter below:
    //   t=0 at l=0  -> kBt[0][c*3+0];   t=2 at l=255 -> kBt[255][c*3+2]
    if (tid < 2 * NP) {
        int c = tid >> 1;
        if (tid & 1) kBt[255 * K2 + c * 3 + 2] = (__bf16)0.0f;
        else         kBt[0   * K2 + c * 3 + 0] = (__bf16)0.0f;
    }

    // ---- Phase 0b (scatter): each k[c][x] feeds 3 tap positions:
    //   Kmat[c*3+t][l] = k[c][l+t-1]  =>  kBt[x+1-t][c*3+t] = k[c][x]
    // Coalesced float4 loads of k, disjoint bf16 LDS scatter (no races).
    {
        const float4* k4 = (const float4*)kb;
        for (int i = tid; i < NP * (HID / 4); i += 256) {
            int c  = i >> 6;
            int x0 = (i & 63) * 4;
            float4 v = k4[i];
            float vv[4] = {v.x, v.y, v.z, v.w};
            int col = c * 3;
#pragma unroll
            for (int e = 0; e < 4; ++e) {
                int x = x0 + e;
                __bf16 bv = (__bf16)vv[e];
                if (x <= 254) kBt[(x + 1) * K2 + col + 0] = bv;   // t=0, l=x+1
                kBt[x * K2 + col + 1] = bv;                        // t=1, l=x
                if (x >= 1)  kBt[(x - 1) * K2 + col + 2] = bv;     // t=2, l=x-1
            }
        }
    }

    // ---- Phase 0c: zero Wmat K-padding columns 304..319 (never written by GEMM1)
    for (int i = tid; i < MPAD * 16; i += 256) {
        int row = i >> 4;
        int col = N1 + (i & 15);
        Wmat[row * K2 + col] = (__bf16)0.0f;
    }

    // ---- Phase 0d: zero LayerNorm accumulators
    if (tid < MPAD) { sums[tid] = 0.f; sumsq[tid] = 0.f; }

    __syncthreads();

    // =======================================================================
    // GEMM1: Wmat(112x304) = fA(112x256) x W_lin(256x304) + b_lin  (bf16 out)
    // 7 x 19 = 133 tiles of 16x16, round-robin over 8 waves, K = 8 steps of 32
    // =======================================================================
    {
        const uint4* wsT4 = (const uint4*)wsT;
        for (int tile = wave; tile < 7 * 19; tile += 8) {
            int mt = tile / 19;
            int nt = tile % 19;
            int Mrow = mt * 16 + ln;
            int o    = nt * 16 + ln;
            v8f c = {};
#pragma unroll
            for (int ks8 = 0; ks8 < 8; ++ks8) {
                int Kb = ks8 * 32;
                FragBf a, bb;
                // A frag: lane-half g holds K = Kb + g*8 + {0..7, 16..23}
                a.u[0] = *(const uint4*)&fA[Mrow * HID + Kb + g * 8];
                a.u[1] = *(const uint4*)&fA[Mrow * HID + Kb + 16 + g * 8];
                // B frag (from L2-resident transposed W_lin):
                // lane-half g holds K = Kb + g*8 + {0..7, 16..23} in wsT row o
                int bi = (o * HID + Kb + 8 * g) >> 3;   // uint4 index
                bb.u[0] = wsT4[bi];
                bb.u[1] = wsT4[bi + 2];
                c = __builtin_amdgcn_wmma_f32_16x16x32_bf16(
                        false, a.v, false, bb.v, (short)0, c, false, false);
            }
            float bias = (o < NP * KS) ? b_lin[o] : 0.f;
#pragma unroll
            for (int r = 0; r < 8; ++r) {
                int row = mt * 16 + r + 8 * g;           // C/D layout: M = r + 8*(lane/16)
                Wmat[row * K2 + o] = (__bf16)(c[r] + bias);
            }
        }
    }
    __syncthreads();

    // =======================================================================
    // GEMM2: out(112x256) = Wmat(112x320) x Kmat(320x256)
    // Each wave owns 2 N-tiles (cols) x 7 M-tiles = 14 accumulators in VGPRs.
    // =======================================================================
    v8f acc[14];
#pragma unroll
    for (int i = 0; i < 14; ++i) acc[i] = (v8f){};

    const int nt0 = wave * 2;
    const int o0  = nt0 * 16 + ln;
    const int o1  = o0 + 16;

    for (int kk = 0; kk < 10; ++kk) {
        int Kb = kk * 32;
        FragBf b0, b1;
        b0.u[0] = *(const uint4*)&kBt[o0 * K2 + Kb + 8 * g];
        b0.u[1] = *(const uint4*)&kBt[o0 * K2 + Kb + 16 + 8 * g];
        b1.u[0] = *(const uint4*)&kBt[o1 * K2 + Kb + 8 * g];
        b1.u[1] = *(const uint4*)&kBt[o1 * K2 + Kb + 16 + 8 * g];
#pragma unroll
        for (int mt = 0; mt < 7; ++mt) {
            FragBf a;
            int Mrow = mt * 16 + ln;
            a.u[0] = *(const uint4*)&Wmat[Mrow * K2 + Kb + g * 8];
            a.u[1] = *(const uint4*)&Wmat[Mrow * K2 + Kb + 16 + g * 8];
            acc[mt * 2 + 0] = __builtin_amdgcn_wmma_f32_16x16x32_bf16(
                false, a.v, false, b0.v, (short)0, acc[mt * 2 + 0], false, false);
            acc[mt * 2 + 1] = __builtin_amdgcn_wmma_f32_16x16x32_bf16(
                false, a.v, false, b1.v, (short)0, acc[mt * 2 + 1], false, false);
        }
    }

    // =======================================================================
    // LayerNorm over last dim (256): row sums via half-wave shuffles + LDS atomics
    // =======================================================================
#pragma unroll
    for (int mt = 0; mt < 7; ++mt) {
#pragma unroll
        for (int r = 0; r < 8; ++r) {
            float v0 = acc[mt * 2 + 0][r];
            float v1 = acc[mt * 2 + 1][r];
            float s = v0 + v1;
            float q = v0 * v0 + v1 * v1;
#pragma unroll
            for (int off = 8; off > 0; off >>= 1) {
                s += __shfl_xor(s, off, 16);
                q += __shfl_xor(q, off, 16);
            }
            if (ln == 0) {
                int row = mt * 16 + r + 8 * g;
                atomicAdd(&sums[row], s);
                atomicAdd(&sumsq[row], q);
            }
        }
    }
    __syncthreads();

    // Normalize + scale/shift + store (rows >= 100 dropped)
    float* ob = out + (size_t)b * NP * HID;
    const float ga0 = gamma[o0], be0 = beta[o0];
    const float ga1 = gamma[o1], be1 = beta[o1];
#pragma unroll
    for (int mt = 0; mt < 7; ++mt) {
#pragma unroll
        for (int r = 0; r < 8; ++r) {
            int row = mt * 16 + r + 8 * g;
            if (row < NP) {
                float mu   = sums[row]  * (1.f / 256.f);
                float var  = sumsq[row] * (1.f / 256.f) - mu * mu;
                float rstd = rsqrtf(var + EPS);
                ob[row * HID + o0] = (acc[mt * 2 + 0][r] - mu) * rstd * ga0 + be0;
                ob[row * HID + o1] = (acc[mt * 2 + 1][r] - mu) * rstd * ga1 + be1;
            }
        }
    }
}

// ---------------------------------------------------------------------------
extern "C" void kernel_launch(void* const* d_in, const int* in_sizes, int n_in,
                              void* d_out, int out_size, void* d_ws, size_t ws_size,
                              hipStream_t stream) {
    const float* f     = (const float*)d_in[0];
    const float* k     = (const float*)d_in[1];
    const float* W_lin = (const float*)d_in[2];
    const float* b_lin = (const float*)d_in[3];
    const float* gamma = (const float*)d_in[4];
    const float* beta  = (const float*)d_in[5];
    float* out = (float*)d_out;
    __bf16* wsT = (__bf16*)d_ws;   // 304*256*2 = 155648 bytes

    (void)in_sizes; (void)n_in; (void)out_size; (void)ws_size;

    // Allow the large dynamic-LDS allocation (idempotent; safe under capture).
    static_assert(SMEM_BYTES <= 320 * 1024, "fits WGP LDS");
    hipFuncSetAttribute((const void*)dyconv_fused,
                        hipFuncAttributeMaxDynamicSharedMemorySize, SMEM_BYTES);

    dyconv_prep_wlinT<<<(N1 * HID + 255) / 256, 256, 0, stream>>>(W_lin, wsT);
    dyconv_fused<<<BATCH, 256, SMEM_BYTES, stream>>>(f, k, b_lin, gamma, beta, wsT, out);
}
